// DagEncoder_7232724927133
// MI455X (gfx1250) — matrix operand order repaired and checked by reading, not compiled, not run on hardware
//
#include <hip/hip_runtime.h>

typedef __attribute__((ext_vector_type(16))) _Float16 v16h;
typedef __attribute__((ext_vector_type(8)))  float    v8f;
typedef __attribute__((ext_vector_type(4)))  float    v4f;

#define F_DIM   16
#define E_DIM   128
#define H_DIM   128
#define IN_DIM  144
#define K1PAD   160
#define TILE_M  128
#define BLOCK   256

// LDS layout (bytes)
#define OFF_FEAT0  0            // [TILE_M][IN_DIM] fp32 = 73728 (double buffered)
#define OFF_FEAT1  73728
#define OFF_SOUT   147456       // [TILE_M][E_DIM] fp32 = 65536
#define OFF_H1     212992       // [TILE_M][H_DIM] f16  = 32768
#define OFF_W1T    245760       // [H_DIM][K1PAD]  f16  = 40960
#define OFF_W2T    286720       // [E_DIM][H_DIM]  f16  = 32768
#define OFF_B1     319488
#define OFF_B2     320000
#define OFF_SEG    320512
#define SMEM_BYTES 321024       // <= 320KB WGP LDS

// async-per-wave issue count per tile: (8KB x-block + 64KB h-block) / (32 lanes * 16B) / 8 waves
#define ASYNC_PER_WAVE 18

#if defined(__HIP_DEVICE_COMPILE__)
#define WAIT_ASYNC(n) asm volatile("s_wait_asynccnt %0" :: "i"(n) : "memory")
#else
#define WAIT_ASYNC(n)
#endif

__device__ __forceinline__ void async_b128(unsigned ldsOff, unsigned gOff, const void* base) {
#if defined(__HIP_DEVICE_COMPILE__)
  // GVS mode: mem_addr = SGPR_U64 + VGPR_I32 + 0 ; LDS dest = per-lane VGPR byte offset
  asm volatile("global_load_async_to_lds_b128 %0, %1, %2"
               :: "v"(ldsOff), "v"(gOff), "s"(base) : "memory");
#endif
}

// Stage one 128-node tile of concat(x, h_node) as raw fp32 into LDS [128][144].
__device__ __forceinline__ void issue_feat(const float* __restrict__ xB,
                                           const float* __restrict__ hB,
                                           long long node0, long long N,
                                           unsigned featLds, int wave, int lane) {
  const long long rem = N - node0;                       // >= 1 by construction
  const long long maxOffX = rem * (F_DIM * 4) - 16;      // clamp for tail tile
  const long long maxOffH = rem * (E_DIM * 4) - 16;
  const char* xb = (const char*)xB + node0 * (F_DIM * 4);
  const char* hb = (const char*)hB + node0 * (E_DIM * 4);
#pragma unroll
  for (int j = 0; j < ASYNC_PER_WAVE; ++j) {
    const int q = wave * ASYNC_PER_WAVE + j;             // 0..143 chunk id
    if (q < 16) {                                        // x block: 16 chunks x 512B
      long long off = (long long)q * 2048 + lane * 16;
      if (off > maxOffX) off = maxOffX;
      const int e = q * 128 + lane * 4;                  // fp32 element in x block
      const int r = e >> 4, c = e & 15;
      async_b128(featLds + (unsigned)(r * (IN_DIM * 4) + c * 4), (unsigned)off, xb);
    } else {                                             // h block: 128 chunks (1 row each)
      const int p = q - 16;
      long long off = (long long)p * 512 + lane * 16;
      if (off > maxOffH) off = maxOffH;
      async_b128(featLds + (unsigned)(p * (IN_DIM * 4) + F_DIM * 4 + lane * 16),
                 (unsigned)off, hb);
    }
  }
}

__global__ void dag_zero_kernel(float* __restrict__ out, long long n) {
  long long i = (long long)blockIdx.x * blockDim.x + threadIdx.x;
  if (i < n) out[i] = 0.f;
}

__global__ __launch_bounds__(BLOCK)
void dag_mlp_seg_kernel(const float* __restrict__ x,
                        const float* __restrict__ hn,
                        const long long* __restrict__ ptr,
                        const float* __restrict__ W1,
                        const float* __restrict__ b1,
                        const float* __restrict__ W2,
                        const float* __restrict__ b2,
                        float* __restrict__ out,
                        long long N, int B, int numTiles)
{
  extern __shared__ char smem[];
  float*    sOut = (float*)   (smem + OFF_SOUT);
  _Float16* sH1  = (_Float16*)(smem + OFF_H1);
  _Float16* sW1t = (_Float16*)(smem + OFF_W1T);
  _Float16* sW2t = (_Float16*)(smem + OFF_W2T);
  float*    sB1  = (float*)   (smem + OFF_B1);
  float*    sB2  = (float*)   (smem + OFF_B2);
  int*      sSeg = (int*)     (smem + OFF_SEG);

  const int tid  = threadIdx.x;
  const int wave = tid >> 5;
  const int lane = tid & 31;
  const int lr   = lane & 15;
  const int g    = lane >> 4;

  const unsigned featLds0 = (unsigned)(size_t)(smem + OFF_FEAT0);
  const unsigned featLds1 = (unsigned)(size_t)(smem + OFF_FEAT1);

  // ---- prologue: kick off async feature load for this WG's first tile ----
  const int tile0 = blockIdx.x;
  if (tile0 < numTiles)
    issue_feat(x, hn, (long long)tile0 * TILE_M, N, featLds0, wave, lane);

  // ---- stage weights once per workgroup (transposed, f16) ----
  for (int idx = tid; idx < IN_DIM * H_DIM; idx += BLOCK) {
    int k = idx >> 7, h = idx & 127;                 // W1[k][h]
    sW1t[h * K1PAD + k] = (_Float16)W1[idx];
  }
  for (int h = tid; h < H_DIM; h += BLOCK)
    for (int k = IN_DIM; k < K1PAD; ++k)
      sW1t[h * K1PAD + k] = (_Float16)0.f;
  for (int idx = tid; idx < H_DIM * E_DIM; idx += BLOCK) {
    int k = idx >> 7, e = idx & 127;                 // W2[k][e]
    sW2t[e * H_DIM + k] = (_Float16)W2[idx];
  }
  if (tid < 128) { sB1[tid] = b1[tid]; sB2[tid] = b2[tid]; }
  __syncthreads();

  int cur = 0;
  for (int tile = tile0; tile < numTiles; tile += gridDim.x) {
    const long long node0 = (long long)tile * TILE_M;

    // ---- software pipeline: start next tile's async copy, wait for current ----
    const int nxt = tile + gridDim.x;
    if (nxt < numTiles) {
      issue_feat(x, hn, (long long)nxt * TILE_M, N, cur ? featLds0 : featLds1, wave, lane);
      WAIT_ASYNC(ASYNC_PER_WAVE);   // in-order: current buffer's loads are complete
    } else {
      WAIT_ASYNC(0);
    }

    // segment ids for current tile (CSR binary search)
    if (tid < TILE_M) {
      long long node = node0 + tid;
      int s = -1;
      if (node < N) {
        int lo = 0, hi = B;                          // largest j: ptr[j] <= node
        while (lo < hi) {
          int mid = (lo + hi + 1) >> 1;
          if (ptr[mid] <= node) lo = mid; else hi = mid - 1;
        }
        s = lo;
      }
      sSeg[tid] = s;
    }
    __syncthreads();   // current feat buffer visible to all waves; sSeg ready

    const float* feat = (const float*)(smem + (cur ? OFF_FEAT1 : OFF_FEAT0));
    const int mbase = wave * 16;

    // ---- layer 1: [16 x 144] @ [144 x 128], WMMA f16 -> f32 accum ----
    v8f acc1[8];
#pragma unroll
    for (int n = 0; n < 8; ++n) {
      float bv = sB1[n * 16 + lr];
      acc1[n] = (v8f){bv, bv, bv, bv, bv, bv, bv, bv};
    }
    const float* aB = feat + (mbase + lr) * IN_DIM;
#pragma unroll
    for (int s = 0; s < 5; ++s) {
      const int k = (s < 4) ? (s * 32 + g * 8) : (128 + g * 8);
      v4f l0 = *(const v4f*)(aB + k);
      v4f l1 = *(const v4f*)(aB + k + 4);
      v4f h0 = {0.f, 0.f, 0.f, 0.f}, h1 = {0.f, 0.f, 0.f, 0.f};
      if (s < 4) {                                    // K-overhang (144..159) stays zero
        h0 = *(const v4f*)(aB + k + 16);
        h1 = *(const v4f*)(aB + k + 20);
      }
      v16h a;
#pragma unroll
      for (int i = 0; i < 4; ++i) {
        a[i]      = (_Float16)l0[i];
        a[i + 4]  = (_Float16)l1[i];
        a[i + 8]  = (_Float16)h0[i];
        a[i + 12] = (_Float16)h1[i];
      }
      const int kB0 = (s < 4) ? (s * 32 + g * 8) : (128 + g * 8);
#pragma unroll
      for (int n = 0; n < 8; ++n) {
        v16h bf;
        const _Float16* bRow = sW1t + (n * 16 + lr) * K1PAD + kB0;
#pragma unroll
        for (int i = 0; i < 8; ++i) {
          bf[i]     = bRow[i];
          bf[i + 8] = bRow[16 + i];
        }
        acc1[n] = __builtin_amdgcn_wmma_f32_16x16x32_f16(
            false, a, false, bf, (short)0, acc1[n], false, false);
      }
    }
    // ReLU, f16, D-layout -> row-major LDS
#pragma unroll
    for (int n = 0; n < 8; ++n)
#pragma unroll
      for (int j = 0; j < 8; ++j) {
        float v = acc1[n][j];
        v = v > 0.f ? v : 0.f;
        sH1[(mbase + g * 8 + j) * H_DIM + n * 16 + lr] = (_Float16)v;
      }
    __syncthreads();

    // ---- layer 2: [16 x 128] @ [128 x 128] ----
    v8f acc2[8];
#pragma unroll
    for (int n = 0; n < 8; ++n) {
      float bv = sB2[n * 16 + lr];
      acc2[n] = (v8f){bv, bv, bv, bv, bv, bv, bv, bv};
    }
    const _Float16* a2B = sH1 + (mbase + lr) * H_DIM;
#pragma unroll
    for (int s = 0; s < 4; ++s) {
      const int kA0 = s * 32 + g * 16;
      v16h a;
#pragma unroll
      for (int i = 0; i < 8; ++i) {
        a[i]     = a2B[kA0 + i];
        a[i + 8] = a2B[kA0 + 8 + i];
      }
      const int kB0 = s * 32 + g * 16;
#pragma unroll
      for (int n = 0; n < 8; ++n) {
        v16h bf = *(const v16h*)(sW2t + (n * 16 + lr) * H_DIM + kB0);
        acc2[n] = __builtin_amdgcn_wmma_f32_16x16x32_f16(
            false, a, false, bf, (short)0, acc2[n], false, false);
      }
    }
    // per-node fp32 outputs to LDS (D layout: vgpr j -> row mbase + 8g + j)
#pragma unroll
    for (int n = 0; n < 8; ++n)
#pragma unroll
      for (int j = 0; j < 8; ++j)
        sOut[(mbase + g * 8 + j) * E_DIM + n * 16 + lr] = acc2[n][j];
    __syncthreads();

    // ---- CSR segment reduction: serial scan, one atomic per segment change ----
    {
      const int col = (wave & 3) * 32 + lane;        // 4 col-blocks x 32 lanes
      const int r0  = (wave >> 2) * 64;              // 2 row halves
      float accv = 0.f;
      int curSeg = -1;
      for (int r = r0; r < r0 + 64; ++r) {
        int s = sSeg[r];
        if (s != curSeg) {
          if (curSeg >= 0) atomicAdd(&out[(size_t)curSeg * E_DIM + col], accv);
          accv = 0.f;
          curSeg = s;
        }
        if (s >= 0) accv += sOut[r * E_DIM + col];
      }
      if (curSeg >= 0) atomicAdd(&out[(size_t)curSeg * E_DIM + col], accv);
    }
    __syncthreads();

    cur ^= 1;
  }
}

extern "C" void kernel_launch(void* const* d_in, const int* in_sizes, int n_in,
                              void* d_out, int out_size, void* d_ws, size_t ws_size,
                              hipStream_t stream) {
  const float*     x   = (const float*)d_in[0];
  const float*     hn  = (const float*)d_in[1];
  const long long* ptr = (const long long*)d_in[2];
  const float*     W1  = (const float*)d_in[3];
  const float*     b1  = (const float*)d_in[4];
  const float*     W2  = (const float*)d_in[5];
  const float*     b2  = (const float*)d_in[6];
  float* out = (float*)d_out;

  const long long N = (long long)in_sizes[0] / F_DIM;
  const int B = in_sizes[2] - 1;
  const int numTiles = (int)((N + TILE_M - 1) / TILE_M);

  // zero output (harness poisons d_out; empty segments must be exactly 0)
  const long long outElems = (long long)out_size;
  int zgrid = (int)((outElems + BLOCK - 1) / BLOCK);
  dag_zero_kernel<<<zgrid, BLOCK, 0, stream>>>(out, outElems);

  int grid = numTiles < 4096 ? numTiles : 4096;
  dag_mlp_seg_kernel<<<grid, BLOCK, SMEM_BYTES, stream>>>(
      x, hn, ptr, W1, b1, W2, b2, out, N, B, numTiles);
}
